// STMetaGRUEncoder_59210419143003
// MI455X (gfx1250) — matrix operand order repaired and checked by reading, not compiled, not run on hardware
//
#include <hip/hip_runtime.h>
#include <hip/hip_bf16.h>
#include <cstddef>
#include <cstdint>

// ---------------------------------------------------------------------------
// STMetaGRUEncoder for MI455X (gfx1250, wave32, WMMA + async-to-LDS)
//
// Sizes: B=16 T=12 N=325 D=1 H=64 META=127 LH=128, BN=5200
// Dominant work: Wh = hid_wh[5200,128] @ wh_w2[128,12288]  (16.4 GFLOP, 256MB out)
//   -> split-bf16 WMMA GEMM (a_hi*b_hi + a_hi*b_lo + a_lo*b_hi), fp32 accumulate
// GRU scan keeps each (b,n) Wh tile (49KB) in LDS across all 12 timesteps so
// Wh is read from HBM exactly once (~22us total Wh traffic at 23.3 TB/s).
// The LDS fill uses GLOBAL_LOAD_ASYNC_TO_LDS_B128 (ASYNCcnt) so no VGPR
// staging / VALU is burned on the bulk copy.
// ---------------------------------------------------------------------------

#define BN_TOT   5200
#define NNODES   325
#define BB       16
#define TT       12
#define HH       64
#define METAD    127
#define LH       128
#define WH_N     12288      // 3*H*H per H row = 192*64
#define GATE_N   192        // 3*H

typedef __attribute__((ext_vector_type(16))) __bf16 v16bf;
typedef __attribute__((ext_vector_type(8)))  float  v8f;

// workspace layout (dword offsets)
#define HID_OFF     ((size_t)0)                       // [3][5200][128] f32
#define BPK_HI_OFF  ((size_t)1996800)                 // [4][768][32][8] u32 (bf16x2)
#define BPK_LO_OFF  ((size_t)2783232)
#define WH_OFF      ((size_t)3569664)                 // [5200][12288] f32

// ---------------------------------------------------------------------------
// Kernel 1: hidden = relu(meta @ w1 + b1) for the 3 hyper-MLPs.
// grid (5200, 3), block 128
// ---------------------------------------------------------------------------
__global__ void mlp_hidden_kernel(const float* __restrict__ meta,
                                  const float* __restrict__ wx_w1, const float* __restrict__ wx_b1,
                                  const float* __restrict__ wh_w1, const float* __restrict__ wh_b1,
                                  const float* __restrict__ b_w1,  const float* __restrict__ b_b1,
                                  float* __restrict__ hid)
{
    __shared__ float m[METAD];
    const int bn = blockIdx.x;
    const int s  = blockIdx.y;
    const int j  = threadIdx.x;
    const float* w1 = (s == 0) ? wx_w1 : ((s == 1) ? wh_w1 : b_w1);
    const float* b1 = (s == 0) ? wx_b1 : ((s == 1) ? wh_b1 : b_b1);
    if (j < METAD) m[j] = meta[(size_t)bn * METAD + j];
    __syncthreads();
    float acc = b1[j];
    #pragma unroll 1
    for (int k = 0; k < METAD; ++k)
        acc = fmaf(m[k], w1[(size_t)k * LH + j], acc);
    hid[((size_t)s * BN_TOT + bn) * LH + j] = fmaxf(acc, 0.0f);
}

// ---------------------------------------------------------------------------
// Kernel 2: prepack wh_w2[128,12288] into bf16 hi/lo B-fragments laid out
// in contiguous k order (k = 16*half + elem), the same order kernel 3 uses for A:
//   frag[ks][ntile][lane][v]: lane<16 -> col=lane,    K = ks*32 + 2v,2v+1
//                             lane>=16 -> col=lane-16, K = ks*32 + 16 + 2v, +1
// 98304 threads total.
// ---------------------------------------------------------------------------
__global__ void prepack_b_kernel(const float* __restrict__ w2,
                                 unsigned int* __restrict__ bhi,
                                 unsigned int* __restrict__ blo)
{
    const int gid   = blockIdx.x * blockDim.x + threadIdx.x;
    const int lane  = gid & 31;
    const int ntile = (gid >> 5) % (WH_N / 16);
    const int ks    = gid / (32 * (WH_N / 16));
    const int col   = ntile * 16 + (lane & 15);
    const int hl    = lane >> 4;
    const size_t base = ((size_t)(ks * (WH_N / 16) + ntile) * 32 + lane) * 8;
    #pragma unroll
    for (int v = 0; v < 8; ++v) {
        const int k0 = ks * 32 + hl * 16 + 2 * v;
        const float f0 = w2[(size_t)k0 * WH_N + col];
        const float f1 = w2[(size_t)(k0 + 1) * WH_N + col];
        const __bf16 h0 = (__bf16)f0, h1 = (__bf16)f1;
        const __bf16 l0 = (__bf16)(f0 - (float)h0);
        const __bf16 l1 = (__bf16)(f1 - (float)h1);
        union { __bf16 h[2]; unsigned int u; } ph, pl;
        ph.h[0] = h0; ph.h[1] = h1;
        pl.h[0] = l0; pl.h[1] = l1;
        bhi[base + v] = ph.u;
        blo[base + v] = pl.u;
    }
}

// ---------------------------------------------------------------------------
// Kernel 3: Wh = hid_wh @ wh_w2 + wh_b2 via split-bf16 WMMA.
// grid (325, 24), block 256 (8 waves). Each wave: 16(M) x 64(N) tile,
// 4 K-steps x 4 N-subtiles x 3 v_wmma_f32_16x16x32_bf16.
// ---------------------------------------------------------------------------
__global__ void wh_gemm_kernel(const float* __restrict__ hid,          // ws hid base
                               const unsigned int* __restrict__ bhi,
                               const unsigned int* __restrict__ blo,
                               const float* __restrict__ wh_b2,
                               float* __restrict__ Wh)
{
    const int lane = threadIdx.x & 31;
    const int w    = threadIdx.x >> 5;          // wave in block: 0..7
    const int hl   = lane >> 4;
    const int lm   = lane & 15;
    const int row0 = blockIdx.x * 16;
    // A = hid slab s=1 (wh MLP hidden), row-major [5200,128]
    const float* Arow = hid + ((size_t)BN_TOT + (size_t)(row0 + lm)) * LH;

    v8f acc[4];
    #pragma unroll
    for (int s2 = 0; s2 < 4; ++s2)
        #pragma unroll
        for (int r = 0; r < 8; ++r) acc[s2][r] = 0.0f;

    #pragma unroll
    for (int ks = 0; ks < 4; ++ks) {
        // ---- A fragment: lane<16 needs K = kb..kb+15 (kb=ks*32),
        //      lane>=16 the same shifted by +16.
        const int kb = ks * 32 + hl * 16;
        const float4 q0 = *(const float4*)(Arow + kb);
        const float4 q1 = *(const float4*)(Arow + kb + 4);
        const float4 q2 = *(const float4*)(Arow + kb + 8);
        const float4 q3 = *(const float4*)(Arow + kb + 12);
        const float av[16] = { q0.x, q0.y, q0.z, q0.w, q1.x, q1.y, q1.z, q1.w,
                               q2.x, q2.y, q2.z, q2.w, q3.x, q3.y, q3.z, q3.w };
        v16bf ahi, alo;
        #pragma unroll
        for (int i = 0; i < 16; ++i) {
            const __bf16 h = (__bf16)av[i];
            ahi[i] = h;
            alo[i] = (__bf16)(av[i] - (float)h);
        }
        #pragma unroll
        for (int sub = 0; sub < 4; ++sub) {
            const int ntile = blockIdx.y * 32 + w * 4 + sub;
            const size_t fidx = ((size_t)(ks * (WH_N / 16) + ntile) * 32 + lane) * 8;
            union { uint4 q[2]; v16bf v; } fh, fl;
            const uint4* ph = (const uint4*)(bhi + fidx);
            const uint4* pl = (const uint4*)(blo + fidx);
            fh.q[0] = ph[0]; fh.q[1] = ph[1];
            fl.q[0] = pl[0]; fl.q[1] = pl[1];
            // D = A_hi*B_hi + A_hi*B_lo + A_lo*B_hi + C   (error-corrected bf16)
            acc[sub] = __builtin_amdgcn_wmma_f32_16x16x32_bf16(
                false, ahi, false, fh.v, (short)0, acc[sub], false, false);
            acc[sub] = __builtin_amdgcn_wmma_f32_16x16x32_bf16(
                false, ahi, false, fl.v, (short)0, acc[sub], false, false);
            acc[sub] = __builtin_amdgcn_wmma_f32_16x16x32_bf16(
                false, alo, false, fh.v, (short)0, acc[sub], false, false);
        }
    }

    // Epilogue: C/D layout -> element (r,lane): M = r + hl*8, N = lm
    #pragma unroll
    for (int sub = 0; sub < 4; ++sub) {
        const int col = blockIdx.y * 512 + w * 64 + sub * 16 + lm;
        const float bias = wh_b2[col];
        const int grow = row0 + hl * 8;
        #pragma unroll
        for (int r = 0; r < 8; ++r)
            Wh[(size_t)(grow + r) * WH_N + col] = acc[sub][r] + bias;
    }
}

// ---------------------------------------------------------------------------
// Kernel 4: GRU scan. One block per (b,n); Wh tile lives in LDS for all T.
// block 192 threads (6 waves): j<64 -> r gate, 64..127 -> z gate, 128..191 -> n.
// Wh tile is DMA'd memory->LDS with GLOBAL_LOAD_ASYNC_TO_LDS_B128 (ASYNCcnt).
// ---------------------------------------------------------------------------
__global__ void __launch_bounds__(GATE_N)
gru_scan_kernel(const float* __restrict__ x,
                const float* __restrict__ hid,     // ws hid base [3][5200][128]
                const float* __restrict__ wx_w2, const float* __restrict__ wx_b2,
                const float* __restrict__ b_w2,  const float* __restrict__ b_b2,
                const float* __restrict__ Wh,      // [5200][12288]
                float* __restrict__ out)           // h_each_step ++ h_last
{
    __shared__ float sWh[HH * GATE_N];   // 49152 B
    __shared__ float sWx[GATE_N];
    __shared__ float sB[GATE_N];
    __shared__ float sHid[2][LH];
    __shared__ float sH[HH];
    __shared__ float sRH[HH];
    __shared__ float sZ[HH];

    const int bn = blockIdx.x;
    const int b  = bn / NNODES;
    const int n  = bn - b * NNODES;
    const int j  = threadIdx.x;

    // ---- bulk tile fill: async memory->LDS, 128-bit granules, coalesced.
    // LDS byte offset of sWh comes from the low 32 bits of its flat address
    // (LDS aperture maps addr[31:0] directly to the LDS offset).
    {
        const unsigned lds0 = (unsigned)(uintptr_t)(&sWh[0]);
        const float* gsrc = Wh + (size_t)bn * WH_N;
        #pragma unroll
        for (int i = 0; i < (HH * GATE_N) / (GATE_N * 4); ++i) {  // 16 iters
            const int e = (i * GATE_N + j) * 4;
            const unsigned ldsa = lds0 + (unsigned)(e * 4);
            const float* ga = gsrc + e;
            asm volatile("global_load_async_to_lds_b128 %0, %1, off"
                         :: "v"(ldsa), "v"(ga) : "memory");
        }
    }

    if (j < LH) {
        // wx hidden (slab 0) and b hidden (slab 2)
        sHid[0][j] = hid[((size_t)0 * BN_TOT + bn) * LH + j];
        sHid[1][j] = hid[((size_t)2 * BN_TOT + bn) * LH + j];
    }

    // drain the async copies, then make LDS visible to all waves
    asm volatile("s_wait_asynccnt 0x0" ::: "memory");
    __syncthreads();

    // per-column Wx and bias of this (b,n): tiny second-layer matvecs
    {
        float aw = wx_b2[j];
        float ab = b_b2[j];
        #pragma unroll 4
        for (int k = 0; k < LH; ++k) {
            aw = fmaf(sHid[0][k], wx_w2[(size_t)k * GATE_N + j], aw);
            ab = fmaf(sHid[1][k], b_w2[(size_t)k * GATE_N + j], ab);
        }
        sWx[j] = aw;
        sB[j]  = ab;
    }
    if (j < HH) sH[j] = 0.0f;
    __syncthreads();

    for (int t = 0; t < TT; ++t) {
        const float xt = x[((size_t)b * TT + t) * NNODES + n];  // D==1
        if (j < 2 * HH) {
            // r (j<64) and z (64<=j<128) gates: h @ Wh[:, j]
            float acc = 0.0f;
            #pragma unroll 8
            for (int k = 0; k < HH; ++k)
                acc = fmaf(sH[k], sWh[k * GATE_N + j], acc);
            const float g = 1.0f / (1.0f + __expf(-(xt * sWx[j] + acc + sB[j])));
            if (j < HH) sRH[j]      = g * sH[j];  // r * h
            else        sZ[j - HH]  = g;          // z
        }
        __syncthreads();
        if (j >= 2 * HH) {
            const int jj = j - 2 * HH;
            // n gate: (r*h) @ Wh[:, 128+jj]
            float acc = 0.0f;
            #pragma unroll 8
            for (int k = 0; k < HH; ++k)
                acc = fmaf(sRH[k], sWh[k * GATE_N + j], acc);
            const float nv = tanhf(xt * sWx[j] + acc + sB[j]);
            const float z  = sZ[jj];
            const float hn = z * nv + (1.0f - z) * sH[jj];
            sH[jj] = hn;
            out[(((size_t)b * TT + t) * NNODES + n) * HH + jj] = hn;
        }
        __syncthreads();
    }
    if (j < HH)
        out[(size_t)BB * TT * NNODES * HH + ((size_t)b * NNODES + n) * HH + j] = sH[j];
}

// ---------------------------------------------------------------------------
extern "C" void kernel_launch(void* const* d_in, const int* in_sizes, int n_in,
                              void* d_out, int out_size, void* d_ws, size_t ws_size,
                              hipStream_t stream) {
    const float* x      = (const float*)d_in[0];
    const float* meta   = (const float*)d_in[1];
    const float* wx_w1  = (const float*)d_in[2];
    const float* wx_b1  = (const float*)d_in[3];
    const float* wx_w2  = (const float*)d_in[4];
    const float* wx_b2  = (const float*)d_in[5];
    const float* wh_w1  = (const float*)d_in[6];
    const float* wh_b1  = (const float*)d_in[7];
    const float* wh_w2  = (const float*)d_in[8];
    const float* wh_b2  = (const float*)d_in[9];
    const float* b_w1   = (const float*)d_in[10];
    const float* b_b1   = (const float*)d_in[11];
    const float* b_w2   = (const float*)d_in[12];
    const float* b_b2   = (const float*)d_in[13];
    float* out = (float*)d_out;

    float*        ws_f = (float*)d_ws;
    unsigned int* ws_u = (unsigned int*)d_ws;

    // 1) hyper-MLP hidden activations (3x [5200,128])
    mlp_hidden_kernel<<<dim3(BN_TOT, 3), LH, 0, stream>>>(
        meta, wx_w1, wx_b1, wh_w1, wh_b1, b_w1, b_b1, ws_f + HID_OFF);

    // 2) prepack wh_w2 into bf16 hi/lo WMMA B-fragments
    prepack_b_kernel<<<(4 * (WH_N / 16) * 32) / 256, 256, 0, stream>>>(
        wh_w2, ws_u + BPK_HI_OFF, ws_u + BPK_LO_OFF);

    // 3) big GEMM: Wh[5200,12288] via split-bf16 WMMA
    wh_gemm_kernel<<<dim3(BN_TOT / 16, WH_N / 512), 256, 0, stream>>>(
        ws_f + HID_OFF, ws_u + BPK_HI_OFF, ws_u + BPK_LO_OFF, wh_b2, ws_f + WH_OFF);

    // 4) fused Wx/b matvecs + 12-step GRU scan with LDS-resident Wh tiles
    gru_scan_kernel<<<BN_TOT, GATE_N, 0, stream>>>(
        x, ws_f + HID_OFF, wx_w2, wx_b2, b_w2, b_b2, ws_f + WH_OFF, out);
}
